// MessageFunction_30030411333811
// MI455X (gfx1250) — matrix operand order repaired and checked by reading, not compiled, not run on hardware
//
#include <hip/hip_runtime.h>
#include <hip/hip_bf16.h>

// Problem constants
#define HID    64
#define EDIM   16
#define NE     30000
#define KDIM   4096          // HIDDEN*HIDDEN
#define EPAD   30080         // NE rounded up to 128
#define MBLK   128           // edges per workgroup (8 waves x 16)
#define IBLK   16            // output dims (i) per workgroup
#define KBLK   128           // K block staged per slab
#define LDSS   136           // padded LDS row stride (bf16 elems) -> no bank conflicts
#define NSLAB  ((KDIM / KBLK) * IBLK)   // 512 (kb,il) slabs per workgroup

typedef __attribute__((ext_vector_type(16))) __bf16 v16bf;
typedef __attribute__((ext_vector_type(8)))  float  v8f;
typedef __attribute__((ext_vector_type(4)))  int    v4i;

#define AS1 __attribute__((address_space(1)))
#define AS3 __attribute__((address_space(3)))

union BF16x16 { uint4 q[2]; v16bf v; };

#if defined(__has_builtin)
#  if __has_builtin(__builtin_amdgcn_global_load_async_to_lds_b128)
#    define HAVE_ASYNC_LDS 1
#  else
#    define HAVE_ASYNC_LDS 0
#  endif
#  if __has_builtin(__builtin_amdgcn_s_wait_asynccnt)
#    define HAVE_WAIT_ASYNC 1
#  else
#    define HAVE_WAIT_ASYNC 0
#  endif
#else
#  define HAVE_ASYNC_LDS 0
#  define HAVE_WAIT_ASYNC 0
#endif

__device__ __forceinline__ void wait_async0() {
#if HAVE_WAIT_ASYNC
    __builtin_amdgcn_s_wait_asynccnt(0);
#else
    asm volatile("s_wait_asynccnt 0x0" ::: "memory");
#endif
}

// ---------------------------------------------------------------------------
// Kernel 1: W2 [4096,4096] f32 -> W2T [col][k] bf16 (tiled transpose via LDS)
// ---------------------------------------------------------------------------
__global__ void k_w2t(const float* __restrict__ W2, __bf16* __restrict__ w2t) {
    __shared__ float tile[32][33];
    const int kt = blockIdx.x * 32, ct = blockIdx.y * 32;
    const int tx = threadIdx.x, ty = threadIdx.y;   // block (32, 8)
#pragma unroll
    for (int j = 0; j < 4; ++j)
        tile[ty + j * 8][tx] = W2[(size_t)(kt + ty + j * 8) * KDIM + ct + tx];
    __syncthreads();
#pragma unroll
    for (int j = 0; j < 4; ++j)
        w2t[(size_t)(ct + ty + j * 8) * KDIM + kt + tx] = (__bf16)tile[tx][ty + j * 8];
}

// ---------------------------------------------------------------------------
// Kernel 2: gather hs[e][j] = h[src[e]][j] (f32, zero-padded rows), and write
//           dst indices as float into the output tail.
// ---------------------------------------------------------------------------
__global__ void k_hs_dst(const float* __restrict__ h, const int* __restrict__ eidx,
                         float* __restrict__ hsg, float* __restrict__ out) {
    const int idx = blockIdx.x * 256 + threadIdx.x;   // over EPAD*64
    const int e = idx >> 6, j = idx & 63;
    float v = 0.f;
    if (e < NE) {
        const int s = eidx[e];
        v = h[(size_t)s * HID + j];
        if (j == 0) out[(size_t)NE * HID + e] = (float)eidx[NE + e];
    }
    hsg[(size_t)e * HID + j] = v;
}

// ---------------------------------------------------------------------------
// Kernel 3: hid[e][k] = relu(edge_attr @ W1 + b1) in bf16, zero-padded rows
// ---------------------------------------------------------------------------
__global__ void k_mlp1(const float* __restrict__ ea, const float* __restrict__ W1,
                       const float* __restrict__ b1, __bf16* __restrict__ hid) {
    const int e = blockIdx.x;
    const int k = blockIdx.y * 256 + threadIdx.x;
    if (e >= NE) { hid[(size_t)e * KDIM + k] = (__bf16)0.f; return; }
    float a[EDIM];
#pragma unroll
    for (int d = 0; d < EDIM; ++d) a[d] = ea[e * EDIM + d];
    float acc = b1[k];
#pragma unroll
    for (int d = 0; d < EDIM; ++d) acc += a[d] * W1[d * KDIM + k];
    acc = fmaxf(acc, 0.f);
    hid[(size_t)e * KDIM + k] = (__bf16)acc;
}

// ---------------------------------------------------------------------------
// Kernel 4 (main): fused  m[e,i] = sum_k hid[e,k] * (sum_j W2[k,64i+j] hs[e,j])
//                         + sum_j b2[64i+j] hs[e,j]
// Double-buffered async global->LDS staging of W2 slabs, WMMA bf16 compute,
// immediate hs-contraction (linearity) into an LDS m-accumulator.
// ---------------------------------------------------------------------------
__launch_bounds__(256, 1)
__global__ void k_fused(const __bf16* __restrict__ hid, const __bf16* __restrict__ w2t,
                        const float* __restrict__ hsg, const float* __restrict__ b2,
                        float* __restrict__ out) {
    __shared__ __bf16 w2_lds[2][64 * LDSS];   // double-buffered 64 cols x 128 k slab
    __shared__ float  m_lds[MBLK * IBLK];     // m accumulator [e][i]

    const int tid  = threadIdx.x;
    const int lane = tid & 31;
    const int wave = tid >> 5;
    const int ebase = blockIdx.x * MBLK;
    const int ibase = blockIdx.y * IBLK;

    // ---- bias init: m_lds[e][i] = sum_j b2[(ibase+i)*64+j] * hs[e][j]
    for (int idx = tid; idx < MBLK * IBLK; idx += 256) {
        const int e = idx >> 4, i = idx & 15;
        const float* bp = b2 + (size_t)(ibase + i) * HID;
        const float* hp = hsg + (size_t)(ebase + e) * HID;
        float acc = 0.f;
#pragma unroll
        for (int j = 0; j < HID; ++j) acc += bp[j] * hp[j];
        m_lds[idx] = acc;
    }

    // ---- preload hs values matching the WMMA D layout:
    //      hsv[r][t] = hs[ e=r+(lane>=16?8:0) ][ j = 16t + (lane&15) ]
    float hsv[8][4];
    {
        const int n   = lane & 15;
        const int ehi = (lane >> 4) << 3;
#pragma unroll
        for (int r = 0; r < 8; ++r) {
            const float* hp = hsg + (size_t)(ebase + wave * 16 + r + ehi) * HID + n;
#pragma unroll
            for (int t = 0; t < 4; ++t) hsv[r][t] = hp[t * 16];
        }
    }

    const int koffA = (lane < 16) ? 0 : 16;   // A-layout: lanes 16-31 hold K=16..31
    const int koffB = (lane < 16) ? 0 : 16;   // B-layout: lanes 16-31 hold K=16..31
    const int rowA  = wave * 16 + (lane & 15);

    BF16x16 A[4];

    // stage W2T slab (kb, il) into one LDS buffer: 1024 x 16B chunks, 4/thread
    auto stage_w2 = [&](int slab, __bf16* dstbuf) {
        const int il = slab & 15, kb = slab >> 4;
        const __bf16* src = w2t + (size_t)(ibase + il) * 64 * KDIM + kb * KBLK;
#pragma unroll
        for (int rep = 0; rep < 4; ++rep) {
            const int idx = tid + rep * 256;
            const int cc = idx >> 4, ch = idx & 15;
            const __bf16* g = src + (size_t)cc * KDIM + ch * 8;
            __bf16* l = dstbuf + cc * LDSS + ch * 8;
#if HAVE_ASYNC_LDS
            __builtin_amdgcn_global_load_async_to_lds_b128(
                (AS1 v4i*)g, (AS3 v4i*)l, 0, 0);
#else
            *(uint4*)l = *(const uint4*)g;
#endif
        }
    };

    // A fragments straight from global (rows are wave-private; chunks stay hot)
    auto load_A = [&](int kb) {
        const __bf16* base = hid + (size_t)(ebase + rowA) * KDIM + kb * KBLK + koffA;
#pragma unroll
        for (int kc = 0; kc < 4; ++kc) {
            A[kc].q[0] = *(const uint4*)(base + kc * 32);
            A[kc].q[1] = *(const uint4*)(base + kc * 32 + 8);
        }
    };

    auto compute = [&](int il, const __bf16* buf) {
        v8f C0 = {}, C1 = {}, C2 = {}, C3 = {};
#pragma unroll
        for (int kc = 0; kc < 4; ++kc) {
            BF16x16 B0, B1, B2, B3;
            const __bf16* bp = buf + (lane & 15) * LDSS + kc * 32 + koffB;
            B0.q[0] = *(const uint4*)(bp);
            B0.q[1] = *(const uint4*)(bp + 8);
            B1.q[0] = *(const uint4*)(bp + 16 * LDSS);
            B1.q[1] = *(const uint4*)(bp + 16 * LDSS + 8);
            B2.q[0] = *(const uint4*)(bp + 32 * LDSS);
            B2.q[1] = *(const uint4*)(bp + 32 * LDSS + 8);
            B3.q[0] = *(const uint4*)(bp + 48 * LDSS);
            B3.q[1] = *(const uint4*)(bp + 48 * LDSS + 8);
            C0 = __builtin_amdgcn_wmma_f32_16x16x32_bf16(false, A[kc].v, false, B0.v, (short)0, C0, false, false);
            C1 = __builtin_amdgcn_wmma_f32_16x16x32_bf16(false, A[kc].v, false, B1.v, (short)0, C1, false, false);
            C2 = __builtin_amdgcn_wmma_f32_16x16x32_bf16(false, A[kc].v, false, B2.v, (short)0, C2, false, false);
            C3 = __builtin_amdgcn_wmma_f32_16x16x32_bf16(false, A[kc].v, false, B3.v, (short)0, C3, false, false);
        }
        // epilogue: contract D tiles with hs, reduce over N (16-lane halves)
        float s[8];
#pragma unroll
        for (int r = 0; r < 8; ++r)
            s[r] = C0[r] * hsv[r][0] + C1[r] * hsv[r][1] + C2[r] * hsv[r][2] + C3[r] * hsv[r][3];
#pragma unroll
        for (int r = 0; r < 8; ++r) {
            s[r] += __shfl_xor(s[r], 1, 32);
            s[r] += __shfl_xor(s[r], 2, 32);
            s[r] += __shfl_xor(s[r], 4, 32);
            s[r] += __shfl_xor(s[r], 8, 32);
        }
        float sel = s[0];
#pragma unroll
        for (int r = 1; r < 8; ++r) sel = ((lane & 7) == r) ? s[r] : sel;
        if ((lane & 8) == 0) {  // lanes 0-7 (e=r) and 16-23 (e=r+8)
            const int e = wave * 16 + (lane & 7) + ((lane >> 4) << 3);
            m_lds[e * IBLK + il] += sel;
        }
    };

    // ---- software-pipelined main loop over 512 (kb,il) slabs
    stage_w2(0, &w2_lds[0][0]);
    wait_async0();
    __syncthreads();            // also publishes m_lds bias init

#pragma unroll 1
    for (int s = 0; s < NSLAB; ++s) {
        if (s + 1 < NSLAB) stage_w2(s + 1, &w2_lds[(s + 1) & 1][0]);
        if ((s & 15) == 0) load_A(s >> 4);
        compute(s & 15, &w2_lds[s & 1][0]);
        wait_async0();          // own async stage of slab s+1 complete
        __syncthreads();        // slab s buffer free + slab s+1 visible to all
    }

    __syncthreads();
    for (int idx = tid; idx < MBLK * IBLK; idx += 256) {
        const int e = idx >> 4, i = idx & 15;
        const int eg = ebase + e;
        if (eg < NE) out[(size_t)eg * HID + ibase + i] = m_lds[idx];
    }
}

// ---------------------------------------------------------------------------
// launcher
// ---------------------------------------------------------------------------
extern "C" void kernel_launch(void* const* d_in, const int* in_sizes, int n_in,
                              void* d_out, int out_size, void* d_ws, size_t ws_size,
                              hipStream_t stream) {
    const float* h   = (const float*)d_in[0];
    const int*   ei  = (const int*)  d_in[1];
    const float* ea  = (const float*)d_in[2];
    const float* W1  = (const float*)d_in[3];
    const float* b1  = (const float*)d_in[4];
    const float* W2  = (const float*)d_in[5];
    const float* b2  = (const float*)d_in[6];
    float* out = (float*)d_out;

    // workspace layout (assumes ws_size >= ~288MB)
    char* ws = (char*)d_ws;
    __bf16* hid = (__bf16*)ws;                                    // EPAD*4096 bf16 = 246.4MB
    __bf16* w2t = (__bf16*)(ws + (size_t)EPAD * KDIM * 2);        // 4096*4096 bf16 = 33.5MB
    float*  hsg = (float*)(ws + (size_t)EPAD * KDIM * 2 + (size_t)KDIM * KDIM * 2); // 7.7MB

    // 1) transpose + convert W2 -> bf16 [col][k]
    k_w2t<<<dim3(KDIM / 32, KDIM / 32), dim3(32, 8), 0, stream>>>(W2, w2t);
    // 2) gather h[src], emit dst tail
    k_hs_dst<<<(EPAD * HID) / 256, 256, 0, stream>>>(h, ei, hsg, out);
    // 3) layer-1 MLP -> bf16 hid
    k_mlp1<<<dim3(EPAD, KDIM / 256), 256, 0, stream>>>(ea, W1, b1, hid);
    // 4) fused WMMA GEMM + hs contraction (async staged, double-buffered)
    k_fused<<<dim3(EPAD / MBLK, HID / IBLK), 256, 0, stream>>>(hid, w2t, hsg, b2, out);
}
